// RNN_90890097918557
// MI455X (gfx1250) — matrix operand order repaired and checked by reading, not compiled, not run on hardware
//
#include <hip/hip_runtime.h>
#include <hip/hip_bf16.h>

// ---------------------------------------------------------------------------
// CDNA5 (gfx1250) bf16 WMMA implementation of:
//   x_proj = inp @ w_ih                      [B*T, H]  (rows r = b*T+t)
//   scan:  h = sigmoid(h @ w_hh + x_proj_t)  (serial over T, persistent WG)
//   hidden = relu(rnn_out @ fc1_w + fc1_b)   (bf16 in ws)
//   out    = sigmoid(hidden @ fc2_w + fc2_b)
// ---------------------------------------------------------------------------

typedef __bf16 v16bf __attribute__((ext_vector_type(16)));
typedef __bf16 v8bf  __attribute__((ext_vector_type(8)));
typedef float  v8f   __attribute__((ext_vector_type(8)));

#define BATCH 32
#define TSTEPS 1024
#define IDIM 64
#define HDIM 512
#define ADIM 16

__device__ __forceinline__ v8f wmma_bf16(v16bf a, v16bf b, v8f c) {
  // D = A(16x32) * B(32x16) + C, f32 accumulate
  return __builtin_amdgcn_wmma_f32_16x16x32_bf16(
      /*neg_a=*/false, a, /*neg_b=*/false, b,
      /*c_mod=*/(short)0, c, /*reuse_a=*/false, /*reuse_b=*/false);
}

__device__ __forceinline__ v16bf cat8(v8bf lo, v8bf hi) {
  return __builtin_shufflevector(lo, hi, 0, 1, 2, 3, 4, 5, 6, 7,
                                 8, 9, 10, 11, 12, 13, 14, 15);
}

__device__ __forceinline__ v8f zero8() {
  v8f z = {0.f, 0.f, 0.f, 0.f, 0.f, 0.f, 0.f, 0.f};
  return z;
}

__device__ __forceinline__ float sigmoidf_dev(float x) {
  return 1.0f / (1.0f + __expf(-x));
}

// ---------------------------------------------------------------------------
// Pack f32 [K,N] weight into bf16 WMMA B-fragment order:
//   frag(nt,kt): lane l<16 -> col nt*16+l, K = kt*32 + (j&7) + (j>=8?16:0)
//                lane l>=16 -> col nt*16+l-16, K = kt*32 + 8 + (j&7) + (j>=8?16:0)
//   dst flat index == ((nt*KT + kt)*32 + lane)*16 + j
// ---------------------------------------------------------------------------
__global__ void pack_b_kernel(const float* __restrict__ src,
                              __bf16* __restrict__ dst, int K, int N,
                              int total) {
  int idx = blockIdx.x * blockDim.x + threadIdx.x;
  if (idx >= total) return;
  int j = idx & 15;
  int lane = (idx >> 4) & 31;
  int kn = idx >> 9;  // nt*KT + kt
  int KT = K >> 5;
  int kt = kn % KT;
  int nt = kn / KT;
  int k = (kt << 5) + ((lane < 16) ? 0 : 8) + (j & 7) + ((j >> 3) << 4);
  int n = (nt << 4) + (lane & 15);
  dst[idx] = (__bf16)src[(size_t)k * N + n];
}

// ---------------------------------------------------------------------------
// Generic row-major GEMM: out[M,N] = act(A[M,K] @ Bpk + bias)
// Block = 256 threads (8 waves), M-block = 32 rows (2 m-tiles).
// NTB = n-tiles per block (grid.y covers N / (16*NTB)).
// ---------------------------------------------------------------------------
template <int NTB>
__global__ __launch_bounds__(256) void gemm_bf16_kernel(
    const void* __restrict__ Asrc, int aIsBf16, int K,
    const v16bf* __restrict__ Bpk, const float* __restrict__ bias,
    float* __restrict__ outF, __bf16* __restrict__ outH, int N, int act) {
  extern __shared__ char smem_raw[];
  __bf16* sA = (__bf16*)smem_raw;
  const int HS = K + 8;  // padded LDS row stride (halfs)
  const int KT = K >> 5;
  const int tid = threadIdx.x;
  const size_t mBase = (size_t)blockIdx.x * 32;

  // ---- stage A tile (32 x K) into LDS as bf16 ----
  {
    const int chunk = K >> 3;  // per-thread contiguous elements
    const int row = tid >> 3;
    const int col = (tid & 7) * chunk;
    if (aIsBf16) {
      const __bf16* Ah = (const __bf16*)Asrc + (mBase + row) * (size_t)K + col;
      for (int i = 0; i < chunk; ++i) sA[row * HS + col + i] = Ah[i];
    } else {
      const float* Af = (const float*)Asrc + (mBase + row) * (size_t)K + col;
      for (int i = 0; i < chunk; ++i) sA[row * HS + col + i] = (__bf16)Af[i];
    }
  }
  __syncthreads();

  constexpr int WPN = 8 / NTB;             // waves per n-tile
  constexpr int TPW = (WPN == 1) ? 2 : 1;  // m-tiles per (active) wave
  const int wave = tid >> 5;
  const int lane = tid & 31;
  const int ntg = blockIdx.y * NTB + wave / WPN;
  const int mtw = wave % WPN;
  const bool active = (TPW == 2) || (mtw < 2);

  const int laneM = lane & 15;
  const int koffb = (lane < 16) ? 0 : 8;

  if (active) {
    v8f acc[TPW];
#pragma unroll
    for (int ai = 0; ai < TPW; ++ai) acc[ai] = zero8();

    for (int kt = 0; kt < KT; ++kt) {
      v16bf bfr = Bpk[((size_t)ntg * KT + kt) * 32 + lane];
#pragma unroll
      for (int ai = 0; ai < TPW; ++ai) {
        const int mt = (TPW == 2) ? ai : mtw;
        const int rowA = (mt << 4) + laneM;
        const int k1 = (kt << 5) + koffb;
        v8bf lo = *(const v8bf*)&sA[rowA * HS + k1];
        v8bf hi = *(const v8bf*)&sA[rowA * HS + k1 + 16];
        acc[ai] = wmma_bf16(cat8(lo, hi), bfr, acc[ai]);
      }
    }

    // ---- epilogue: bias + activation + store ----
    const int colg = (ntg << 4) + laneM;
    const float bb = bias ? bias[colg] : 0.0f;
    const int rsel = (lane < 16) ? 0 : 8;
#pragma unroll
    for (int ai = 0; ai < TPW; ++ai) {
      const int mt = (TPW == 2) ? ai : mtw;
#pragma unroll
      for (int r = 0; r < 8; ++r) {
        const size_t rowg = mBase + (mt << 4) + rsel + r;
        float v = acc[ai][r] + bb;
        if (act == 1)
          v = fmaxf(v, 0.0f);
        else if (act == 2)
          v = sigmoidf_dev(v);
        const size_t o = rowg * (size_t)N + colg;
        if (outF)
          outF[o] = v;
        else
          outH[o] = (__bf16)v;
      }
    }
  }
}

// ---------------------------------------------------------------------------
// Persistent single-workgroup RNN scan.
// 512 threads = 16 waves (4 waves/SIMD -> full VGPR budget, no spills).
// Wave w owns 2 N-tiles (cols 32w..32w+31):
//   - first RESKT k-tiles of w_hh B-fragments register-resident
//     (2 * RESKT * 8 = 128 VGPRs), remaining k-tiles streamed from L2.
//   - A-fragments of h read from LDS (padded stride, 16B-aligned loads),
//     each A-fragment reused for both owned n-tiles.
// Two barriers per step: compute-reads / h-update.
// ---------------------------------------------------------------------------
#define HLDS_STRIDE (HDIM + 8)
#define SCAN_THREADS 512
#define RESKT 8

__global__ __launch_bounds__(SCAN_THREADS) void rnn_scan_kernel(
    const float* __restrict__ hn, const float* __restrict__ xproj,
    const v16bf* __restrict__ whh_pk, float* __restrict__ rnn_out,
    float* __restrict__ hn_last) {
  __shared__ __bf16 hlds[BATCH * HLDS_STRIDE];

  const int tid = threadIdx.x;
  const int lane = tid & 31;
  const int w = tid >> 5;  // wave id 0..15; owns n-tiles 2w, 2w+1

  // init h0 = hn[0]  (B x H)
  for (int i = tid; i < BATCH * HDIM; i += SCAN_THREADS) {
    const int b = i >> 9;
    const int c = i & (HDIM - 1);
    hlds[b * HLDS_STRIDE + c] = (__bf16)hn[i];
  }

  // register-resident partial B-chain (k-tiles 0..RESKT-1, both n-tiles)
  v16bf Bres[2][RESKT];
#pragma unroll
  for (int ntl = 0; ntl < 2; ++ntl)
#pragma unroll
    for (int kt = 0; kt < RESKT; ++kt)
      Bres[ntl][kt] =
          whh_pk[((size_t)((2 * w + ntl) * 16 + kt)) * 32 + lane];

  __syncthreads();

  const int laneM = lane & 15;
  const int rsel = (lane < 16) ? 0 : 8;
  const int koffb = (lane < 16) ? 0 : 8;

  for (int t = 0; t < TSTEPS; ++t) {
    // prefetch next step's x_proj slab for this thread
    if (t + 1 < TSTEPS)
      __builtin_prefetch(
          &xproj[((size_t)rsel * TSTEPS + (t + 1)) * HDIM + (w << 5) + laneM]);

    v8f acc[2][2];  // [m-tile][local n-tile]
    acc[0][0] = zero8();
    acc[0][1] = zero8();
    acc[1][0] = zero8();
    acc[1][1] = zero8();

#pragma unroll
    for (int kt = 0; kt < 16; ++kt) {
      const int k1 = (kt << 5) + koffb;
      // A-fragments for both m-tiles (shared across the two n-tiles)
      v8bf lo0 = *(const v8bf*)&hlds[(0 + laneM) * HLDS_STRIDE + k1];
      v8bf hi0 = *(const v8bf*)&hlds[(0 + laneM) * HLDS_STRIDE + k1 + 16];
      v8bf lo1 = *(const v8bf*)&hlds[(16 + laneM) * HLDS_STRIDE + k1];
      v8bf hi1 = *(const v8bf*)&hlds[(16 + laneM) * HLDS_STRIDE + k1 + 16];
      v16bf a0 = cat8(lo0, hi0);
      v16bf a1 = cat8(lo1, hi1);
#pragma unroll
      for (int ntl = 0; ntl < 2; ++ntl) {
        v16bf bfr;
        if (kt < RESKT)
          bfr = Bres[ntl][kt & (RESKT - 1)];
        else
          bfr = whh_pk[((size_t)((2 * w + ntl) * 16 + kt)) * 32 + lane];
        acc[0][ntl] = wmma_bf16(a0, bfr, acc[0][ntl]);
        acc[1][ntl] = wmma_bf16(a1, bfr, acc[1][ntl]);
      }
    }

    __syncthreads();  // all reads of h done before overwrite

#pragma unroll
    for (int ntl = 0; ntl < 2; ++ntl) {
      const int colg = (w << 5) + (ntl << 4) + laneM;
#pragma unroll
      for (int mt = 0; mt < 2; ++mt) {
#pragma unroll
        for (int r = 0; r < 8; ++r) {
          const int b = (mt << 4) + rsel + r;
          const size_t idx = ((size_t)b * TSTEPS + t) * HDIM + colg;
          float v = acc[mt][ntl][r] + xproj[idx];
          v = sigmoidf_dev(v);
          rnn_out[idx] = v;
          if (t == TSTEPS - 1) hn_last[b * HDIM + colg] = v;
          hlds[b * HLDS_STRIDE + colg] = (__bf16)v;
        }
      }
    }

    __syncthreads();  // h fully updated before next step's reads
  }
}

// ---------------------------------------------------------------------------
// Host launcher
// ---------------------------------------------------------------------------
extern "C" void kernel_launch(void* const* d_in, const int* in_sizes, int n_in,
                              void* d_out, int out_size, void* d_ws,
                              size_t ws_size, hipStream_t stream) {
  (void)in_sizes;
  (void)n_in;
  (void)out_size;
  (void)ws_size;

  const float* inp = (const float*)d_in[0];    // [B,T,I]
  const float* hn = (const float*)d_in[1];     // [1,B,H]
  const float* w_hh = (const float*)d_in[2];   // [H,H]
  const float* w_ih = (const float*)d_in[3];   // [I,H]
  const float* fc1_w = (const float*)d_in[4];  // [H,H]
  const float* fc1_b = (const float*)d_in[5];  // [H]
  const float* fc2_w = (const float*)d_in[6];  // [H,A]
  const float* fc2_b = (const float*)d_in[7];  // [A]

  float* out = (float*)d_out;
  float* outA = out;                                 // [B,T,A] 524288
  float* hnL = out + (size_t)BATCH * TSTEPS * ADIM;  // [1,B,H] 16384
  float* rnn = hnL + (size_t)BATCH * HDIM;           // [B,T,H] 16777216

  char* ws = (char*)d_ws;
  float* xproj = (float*)ws;                            // 64 MB f32
  __bf16* hidden = (__bf16*)(ws + ((size_t)64 << 20));  // 32 MB bf16
  __bf16* wih_pk = (__bf16*)(ws + ((size_t)96 << 20));  // 64 KB
  __bf16* whh_pk = wih_pk + (size_t)IDIM * HDIM;        // 512 KB
  __bf16* fc1_pk = whh_pk + (size_t)HDIM * HDIM;        // 512 KB
  __bf16* fc2_pk = fc1_pk + (size_t)HDIM * HDIM;        // 16 KB

  // ---- pack weights into WMMA B-fragment order (bf16) ----
  pack_b_kernel<<<(IDIM * HDIM) / 256, 256, 0, stream>>>(w_ih, wih_pk, IDIM,
                                                         HDIM, IDIM * HDIM);
  pack_b_kernel<<<(HDIM * HDIM) / 256, 256, 0, stream>>>(w_hh, whh_pk, HDIM,
                                                         HDIM, HDIM * HDIM);
  pack_b_kernel<<<(HDIM * HDIM) / 256, 256, 0, stream>>>(fc1_w, fc1_pk, HDIM,
                                                         HDIM, HDIM * HDIM);
  pack_b_kernel<<<(HDIM * ADIM) / 256, 256, 0, stream>>>(fc2_w, fc2_pk, HDIM,
                                                         ADIM, HDIM * ADIM);

  const int Mrows = BATCH * TSTEPS;  // 32768

  // ---- x_proj = inp @ w_ih : M=32768, K=64, N=512 (f32 out) ----
  gemm_bf16_kernel<8><<<dim3(Mrows / 32, HDIM / 128), 256,
                        32 * (IDIM + 8) * 2, stream>>>(
      inp, /*aIsBf16=*/0, IDIM, (const v16bf*)wih_pk, nullptr, xproj, nullptr,
      HDIM, /*act=*/0);

  // ---- serial recurrence (persistent single workgroup, 16 waves) ----
  rnn_scan_kernel<<<1, SCAN_THREADS, 0, stream>>>(
      hn, xproj, (const v16bf*)whh_pk, rnn, hnL);

  // ---- hidden = relu(rnn_out @ fc1_w + fc1_b) : bf16 out ----
  gemm_bf16_kernel<8><<<dim3(Mrows / 32, HDIM / 128), 256,
                        32 * (HDIM + 8) * 2, stream>>>(
      rnn, /*aIsBf16=*/0, HDIM, (const v16bf*)fc1_pk, fc1_b, nullptr, hidden,
      HDIM, /*act=*/1);

  // ---- out = sigmoid(hidden @ fc2_w + fc2_b) : M=32768, K=512, N=16 ----
  gemm_bf16_kernel<1><<<dim3(Mrows / 32, 1), 256, 32 * (HDIM + 8) * 2,
                        stream>>>(hidden, /*aIsBf16=*/1, HDIM,
                                  (const v16bf*)fc2_pk, fc2_b, outA, nullptr,
                                  ADIM, /*act=*/2);
}
